// ConcentratedGATv2Layer_42039139893269
// MI455X (gfx1250) — matrix operand order repaired and checked by reading, not compiled, not run on hardware
//
#include <hip/hip_runtime.h>
#include <hip/hip_bf16.h>

// ---------------- problem constants ----------------
#define BDIM     8
#define NNODES   512
#define IN_DIM   128
#define UNITS    64
#define HEADS    4
#define TOPK     32
#define NEGV     (-1.0e9f)
#define ROWS_TOT (BDIM * NNODES)          // 4096 feature rows

// TDM staging: 2 tiles of 256 rows, row stride padded 64->68 dwords
#define TILE_ROWS   256
#define ROW_STRIDE  68                    // 64 data dwords + 4 pad dwords
#define TILE_DW     (TILE_ROWS * ROW_STRIDE)
#define STAGE_BYTES (2 * TILE_DW * 4)     // 139,264 B dynamic LDS

typedef __attribute__((ext_vector_type(8)))  float        v8f;
typedef __attribute__((ext_vector_type(16))) __bf16       v16bf;
typedef __attribute__((ext_vector_type(4)))  unsigned int ui32x4;
typedef __attribute__((ext_vector_type(8)))  int          i32x8;
typedef __attribute__((ext_vector_type(4)))  int          i32x4;

// =====================================================================
// Kernel 1: projections  h = features @ W[h]  for all heads, src & dst.
// bf16 WMMA 16x16x32, fp32 accumulate. One wave computes a 16x64 tile.
// =====================================================================
#define CVT4(dst, o, q)                                   \
  dst[(o)+0] = (__bf16)(q).x; dst[(o)+1] = (__bf16)(q).y; \
  dst[(o)+2] = (__bf16)(q).z; dst[(o)+3] = (__bf16)(q).w;

__global__ __launch_bounds__(256)
void gat_proj_wmma(const float* __restrict__ feat,
                   const float* __restrict__ Wsrc,
                   const float* __restrict__ Wdst,
                   float* __restrict__ hs,
                   float* __restrict__ hd)
{
  const int lane   = threadIdx.x & 31;
  const int wave   = threadIdx.x >> 5;
  const int h      = blockIdx.y;
  const int isDst  = blockIdx.z;
  const int laneHi = lane >> 4;
  const int m      = lane & 15;
  const int rowBase = blockIdx.x * 128 + wave * 16;

  const float* F   = feat + (size_t)(rowBase + m) * IN_DIM;
  const float* Wm  = (isDst ? Wdst : Wsrc) + (size_t)h * IN_DIM * UNITS;
  float*       dst = (isDst ? hd : hs) + (size_t)h * (ROWS_TOT * UNITS);

  v8f acc[4] = {v8f{}, v8f{}, v8f{}, v8f{}};

  #pragma unroll
  for (int k0 = 0; k0 < IN_DIM; k0 += 32) {
    const int cA = k0 + (laneHi ? 8 : 0);
    float4 p0 = *reinterpret_cast<const float4*>(F + cA);
    float4 p1 = *reinterpret_cast<const float4*>(F + cA + 4);
    float4 p2 = *reinterpret_cast<const float4*>(F + cA + 16);
    float4 p3 = *reinterpret_cast<const float4*>(F + cA + 20);
    v16bf afrag;
    CVT4(afrag, 0, p0); CVT4(afrag, 4, p1);
    CVT4(afrag, 8, p2); CVT4(afrag, 12, p3);

    const float* wrow = Wm + (size_t)(k0 + 8 * laneHi) * UNITS + m;
    #pragma unroll
    for (int nt = 0; nt < 4; ++nt) {
      const float* wp = wrow + nt * 16;
      v16bf bfrag;
      #pragma unroll
      for (int s = 0; s < 8; ++s) {
        bfrag[s]     = (__bf16)wp[(size_t)s * UNITS];
        bfrag[8 + s] = (__bf16)wp[(size_t)(16 + s) * UNITS];
      }

      acc[nt] = __builtin_amdgcn_wmma_f32_16x16x32_bf16(
          false, afrag, false, bfrag, (short)0, acc[nt], false, false);
    }
  }

  const int orow = rowBase + 8 * laneHi;
  #pragma unroll
  for (int nt = 0; nt < 4; ++nt)
    #pragma unroll
    for (int r = 0; r < 8; ++r)
      dst[(size_t)(orow + r) * UNITS + nt * 16 + m] = acc[nt][r];
}

// =====================================================================
// TDM: DMA one 2-D tile (rows x 64 f32) from global into LDS with a
// 4-dword pad after every 64-dword row (bank-conflict avoidance, keeps
// 16B alignment for ds_load_b128). Descriptor per ISA 08_async_tensor.
// 6-arg builtin form (clang-23 / therock-10.0 headers).
// =====================================================================
__device__ __forceinline__ void tdm_load_rows(const float* gsrc, unsigned ldsOff,
                                              int rows) {
  const unsigned long long ga = (unsigned long long)(uintptr_t)gsrc;
  ui32x4 g0;
  g0[0] = 1u;                                            // count=1 (valid), user mode
  g0[1] = ldsOff;                                        // lds_addr (bytes)
  g0[2] = (unsigned)(ga & 0xFFFFFFFFu);                  // global_addr[31:0]
  g0[3] = (unsigned)((ga >> 32) & 0x01FFFFFFu) | (2u << 30); // addr[56:32] | type=2
  i32x8 g1;
  g1[0] = (int)((2u << 16)      // data_size = 4 bytes
              | (1u << 20)      // pad_enable
              | (5u << 22)      // pad_interval: every 64 dwords
              | (3u << 25));    // pad_amount: 4 dwords
  g1[1] = (int)(64u << 16);     // tensor_dim0 = 64 (elements per row)
  g1[2] = (int)((unsigned)NNODES << 16);  // tensor_dim1 = 512 rows
  g1[3] = (int)(64u << 16);     // tile_dim0 = 64
  g1[4] = (int)(rows & 0xFFFF); // tile_dim1 = rows, tile_dim2 = 0
  g1[5] = 64;                   // tensor_dim0_stride = 64 elements
  g1[6] = 0;
  g1[7] = 0;
  i32x4 z4 = {0, 0, 0, 0};                  // groups 2/3: zero (2-D tensor)
  i32x8 z8 = {0, 0, 0, 0, 0, 0, 0, 0};
  __builtin_amdgcn_tensor_load_to_lds(g0, g1, z4, z4, z8, 0);
}

// =====================================================================
// Kernel 2: per-row scores -> top-32 -> sparsemax -> sharpen -> sparse
// aggregate; 4 heads looped inside one block per (b, i). h_dst tiles are
// TDM double-buffered into LDS so the VALU-bound score loop never waits
// on global memory.
// =====================================================================
__device__ __forceinline__ float blockReduceSum(float v, float* red) {
  const int t = threadIdx.x;
  red[t] = v;
  __syncthreads();
  #pragma unroll
  for (int s = 128; s > 0; s >>= 1) {
    if (t < s) red[t] += red[t + s];
    __syncthreads();
  }
  float r = red[0];
  __syncthreads();
  return r;
}

__device__ __forceinline__ void bitonicDesc512(float* a, int t) {
  for (int k = 2; k <= NNODES; k <<= 1) {
    for (int jj = k >> 1; jj > 0; jj >>= 1) {
      __syncthreads();
      #pragma unroll
      for (int base = 0; base < NNODES; base += 256) {
        const int ii  = base + t;
        const int ixj = ii ^ jj;
        if (ixj > ii) {
          float x = a[ii], y = a[ixj];
          const bool descBlock = ((ii & k) == 0);
          if (descBlock ? (x < y) : (x > y)) { a[ii] = y; a[ixj] = x; }
        }
      }
    }
  }
  __syncthreads();
}

__global__ __launch_bounds__(256)
void gat_attn(const float* __restrict__ hs,
              const float* __restrict__ hd,
              const int*   __restrict__ adj,
              const float* __restrict__ avec,
              float*       __restrict__ out)
{
  extern __shared__ float hdT[];   // 2 x TILE_DW staged h_dst tiles (TDM)

  __shared__ float sc[NNODES];     // scores -> logits -> z
  __shared__ float zs[NNODES];     // sort buffer
  __shared__ float cA[NNODES];     // scan ping / p values
  __shared__ float cB[NNODES];     // scan pong
  __shared__ float wgt[NNODES];    // compacted normalized attn
  __shared__ int   idxl[NNODES];   // compacted indices
  __shared__ int   validM[NNODES]; // adjacency||self mask
  __shared__ float hsI[UNITS];
  __shared__ float aH[UNITS];
  __shared__ float red[256];
  __shared__ int   cnt;

  const int t     = threadIdx.x;
  const int wave0 = (threadIdx.x >> 5) == 0;
  const int i     = blockIdx.x;
  const int b     = blockIdx.y;

  const unsigned ldsBase = (unsigned)(uintptr_t)&hdT[0];

  // mask is head-independent: build once
  const int* arow = adj + ((size_t)b * NNODES + i) * NNODES;
  for (int j = t; j < NNODES; j += 256)
    validM[j] = (arow[j] != 0) || (j == i);
  __syncthreads();

  float meanAcc = 0.f;

  for (int h = 0; h < HEADS; ++h) {
    const float* HS = hs + (size_t)h * (ROWS_TOT * UNITS) + (size_t)b * NNODES * UNITS;
    const float* HD = hd + (size_t)h * (ROWS_TOT * UNITS) + (size_t)b * NNODES * UNITS;

    // ---- kick off both h_dst tile DMAs (wave 0 only; TENSORcnt-tracked)
    if (wave0) {
      tdm_load_rows(HD,                         ldsBase,               TILE_ROWS);
      tdm_load_rows(HD + (size_t)TILE_ROWS * UNITS,
                                                ldsBase + TILE_DW * 4, TILE_ROWS);
    }

    if (t < UNITS) { hsI[t] = HS[(size_t)i * UNITS + t]; aH[t] = avec[h * UNITS + t]; }

    // ---- scores from LDS: s = 0.125 * sum_u a[u]*leakyrelu(hs_i[u]+hd_j[u])
    #pragma unroll
    for (int tile = 0; tile < 2; ++tile) {
      if (wave0) {
        if (tile == 0) __builtin_amdgcn_s_wait_tensorcnt(1);
        else           __builtin_amdgcn_s_wait_tensorcnt(0);
      }
      __syncthreads();                      // tile data visible to all waves

      const int   j   = tile * TILE_ROWS + t;
      const float* hdr = hdT + (size_t)tile * TILE_DW + (size_t)t * ROW_STRIDE;
      float s = 0.f;
      #pragma unroll
      for (int u = 0; u < UNITS; u += 4) {
        float4 q = *reinterpret_cast<const float4*>(hdr + u);
        float x0 = hsI[u + 0] + q.x; x0 = fmaxf(x0, 0.2f * x0);
        float x1 = hsI[u + 1] + q.y; x1 = fmaxf(x1, 0.2f * x1);
        float x2 = hsI[u + 2] + q.z; x2 = fmaxf(x2, 0.2f * x2);
        float x3 = hsI[u + 3] + q.w; x3 = fmaxf(x3, 0.2f * x3);
        s = fmaf(aH[u + 0], x0, s);
        s = fmaf(aH[u + 1], x1, s);
        s = fmaf(aH[u + 2], x2, s);
        s = fmaf(aH[u + 3], x3, s);
      }
      sc[j] = validM[j] ? s * 0.125f : NEGV;   // 1/sqrt(UNITS) = 0.125
    }
    __syncthreads();

    // ---- top-32 threshold via full descending sort
    for (int j = t; j < NNODES; j += 256) zs[j] = sc[j];
    bitonicDesc512(zs, t);
    const float kth = zs[TOPK - 1];

    // ---- top-k mask + temperature (T=0.5 -> *2)
    for (int j = t; j < NNODES; j += 256) {
      float v = sc[j];
      v = (v >= kth) ? v : NEGV;
      sc[j] = v * 2.0f;
    }
    __syncthreads();

    // ---- sparsemax: valid mean
    float lc = 0.f, ls = 0.f;
    for (int j = t; j < NNODES; j += 256) {
      float v = sc[j];
      if (v > -1.0e8f) { lc += 1.f; ls += v; }
    }
    const float nValid   = blockReduceSum(lc, red);
    const float sumValid = blockReduceSum(ls, red);
    const float mean     = sumValid / fmaxf(nValid, 1.f);

    for (int j = t; j < NNODES; j += 256) {
      float v = sc[j];
      v = (v > -1.0e8f) ? (v - mean) : NEGV;
      sc[j] = v;
      zs[j] = v;
    }
    bitonicDesc512(zs, t);

    // ---- inclusive cumsum of sorted z (Hillis-Steele, ping-pong)
    for (int j = t; j < NNODES; j += 256) cA[j] = zs[j];
    __syncthreads();
    float* src = cA; float* dstb = cB;
    for (int off = 1; off < NNODES; off <<= 1) {
      for (int j = t; j < NNODES; j += 256)
        dstb[j] = src[j] + (j >= off ? src[j - off] : 0.f);
      __syncthreads();
      float* tmp = src; src = dstb; dstb = tmp;
    }

    // ---- support, k*, tau
    float lsupC = 0.f, lsupS = 0.f;
    for (int j = t; j < NNODES; j += 256) {
      const float thr = (src[j] - 1.f) / (float)(j + 1);
      if (zs[j] > thr) { lsupC += 1.f; lsupS += zs[j]; }
    }
    const float kStar = fmaxf(blockReduceSum(lsupC, red), 1.f);
    const float supS  = blockReduceSum(lsupS, red);
    const float tau   = (supS - 1.f) / kStar;

    // ---- attn, sharpen, compact support set (LDS atomic append)
    if (t == 0) cnt = 0;
    __syncthreads();
    float lp = 0.f;
    for (int j = t; j < NNODES; j += 256) {
      const float v   = sc[j];
      const float att = (v > -1.0e8f) ? fmaxf(v - tau, 0.f) : 0.f;
      float p = att + 1e-9f; p = p * p;
      cA[j] = p;          // scan buffers are dead now
      lp += p;
      if (att > 0.f) { int pos = atomicAdd(&cnt, 1); idxl[pos] = j; }
    }
    const float sumP = blockReduceSum(lp, red) + 1e-9f;   // syncs inside
    const int nsel = cnt;
    for (int k = t; k < nsel; k += 256) wgt[k] = cA[idxl[k]] / sumP;
    __syncthreads();

    // ---- sparse aggregate: out_h[i,u] = sum_sel w * h_src[j,u]
    if (t < UNITS) {
      float acc = 0.f;
      for (int k = 0; k < nsel; ++k)
        acc = fmaf(wgt[k], HS[(size_t)idxl[k] * UNITS + t], acc);
      meanAcc += 0.25f * acc;   // mean over 4 heads
    }
    __syncthreads();            // also guards LDS tiles before next head's TDM
  }

  if (t < UNITS)
    out[((size_t)b * NNODES + i) * UNITS + t] = meanAcc;
}

// =====================================================================
extern "C" void kernel_launch(void* const* d_in, const int* in_sizes, int n_in,
                              void* d_out, int out_size, void* d_ws, size_t ws_size,
                              hipStream_t stream) {
  (void)in_sizes; (void)n_in; (void)out_size; (void)ws_size;
  const float* features  = (const float*)d_in[0];
  const int*   adjacency = (const int*)  d_in[1];
  const float* W_src     = (const float*)d_in[2];
  const float* W_dst     = (const float*)d_in[3];
  const float* a         = (const float*)d_in[4];
  float*       out       = (float*)d_out;

  float* hs = (float*)d_ws;                                  // HEADS*4096*64
  float* hd = hs + (size_t)HEADS * ROWS_TOT * UNITS;         // HEADS*4096*64

  dim3 gp(ROWS_TOT / 128, HEADS, 2);
  gat_proj_wmma<<<gp, 256, 0, stream>>>(features, W_src, W_dst, hs, hd);

  dim3 ga(NNODES, BDIM);
  gat_attn<<<ga, 256, STAGE_BYTES, stream>>>(hs, hd, adjacency, a, out);
}
